// custom_v1_25872882991784
// MI455X (gfx1250) — matrix operand order repaired and checked by reading, not compiled, not run on hardware
//
#include <hip/hip_runtime.h>

// Problem: B=256, T=64, S=16, U=4.
//   per-pixel GEMM: Y_p(256x1024) = X_p(256x64) * W_p^T + b_p, then pixel-shuffle.
// Roofline: 8.6 GFLOP vs 353 MB moved (268 MB output) -> memory bound at ~15us
// on 23.3 TB/s HBM if we use the f16 WMMA path (f32 accumulate). Inputs (84 MB)
// fit in the 192 MB L2, so cross-block re-reads of W / xh are L2 hits.
//
// Kernel 1: transpose+convert x (b,t,p) f32 -> xh[p][b*64+t] f16 in d_ws (8 MiB).
// Kernel 2: WMMA GEMM, one block per (pixel, 64-batch M-block, 128-out N-block).
//           X tile staged with gfx1250 async global->LDS b128 (ASYNCcnt),
//           overlapping the W f32->f16 conversion.

typedef __attribute__((ext_vector_type(16))) _Float16 v16h;
typedef __attribute__((ext_vector_type(8)))  float    v8f;
typedef __attribute__((ext_vector_type(8)))  _Float16 h8;

#define XROW 72   // LDS row stride (halves) for X tile: 64 data + 8 pad (16B)
#define WROW 72   // LDS row stride (halves) for W tile

// ---------------------------------------------------------------------------
// Kernel 1: view x as a (BT=16384 rows, P=256 cols) f32 matrix; emit the
// (P, BT) f16 transpose via a 32x32 LDS tile. Reads and writes coalesced.
// ---------------------------------------------------------------------------
__global__ void ups_transpose_cvt(const float* __restrict__ x,
                                  _Float16* __restrict__ xh) {
  __shared__ _Float16 tile[32][33];
  const int tx = threadIdx.x;          // 0..31
  const int ty = threadIdx.y;          // 0..7
  const int bt0 = blockIdx.x * 32;     // row block in (b*T+t)
  const int p0  = blockIdx.y * 32;     // col block in pixel index

#pragma unroll
  for (int i = 0; i < 4; ++i) {
    int r = ty + i * 8;
    tile[r][tx] = (_Float16)x[(size_t)(bt0 + r) * 256 + (p0 + tx)];
  }
  __syncthreads();
#pragma unroll
  for (int i = 0; i < 4; ++i) {
    int pl = ty + i * 8;
    xh[(size_t)(p0 + pl) * 16384 + (bt0 + tx)] = tile[tx][pl];
  }
}

// ---------------------------------------------------------------------------
// Kernel 2: grid (nblk=8, mblk=4, p=256), 256 threads = 8 waves.
// Wave w: M-tile mt = w/2 (16 batches), N-tiles ntb..ntb+3 (ntb = (w&1)*4).
// K=64 -> two v_wmma_f32_16x16x32_f16 per tile, bias pre-loaded into C.
// ---------------------------------------------------------------------------
__global__ void __launch_bounds__(256)
ups_wmma_gemm(const _Float16* __restrict__ xh, const float* __restrict__ W,
              const float* __restrict__ bvec, float* __restrict__ out) {
  __shared__ _Float16 Xs[64 * XROW];    //  9.2 KB
  __shared__ _Float16 Ws[128 * WROW];   // 18.4 KB
  __shared__ float    Bs[128];

  const int nblk = blockIdx.x;          // 0..7  (128 outputs each)
  const int mblk = blockIdx.y;          // 0..3  (64 batches each)
  const int p    = blockIdx.z;          // 0..255
  const int Si = p >> 4, Sj = p & 15;

  const int tid  = threadIdx.x;
  const int lane = tid & 31;
  const int wave = tid >> 5;
  const int half = lane >> 4;           // lane group (ISA K-split)
  const int l16  = lane & 15;

  // ---- stage X block: 4096 contiguous halves from xh. Pure copy -> use
  // gfx1250 async global->LDS (no VGPR round trip, tracked by ASYNCcnt;
  // the instruction offset advances BOTH the global and LDS address, so the
  // second b128 lands at +16B in our padded LDS row automatically).
  {
    const _Float16* src = xh + (size_t)p * (256 * 64) + (size_t)mblk * (64 * 64);
    const int h0 = tid * 16;                 // 16 halves (32B) / thread
    const int r = h0 >> 6, c = h0 & 63;      // c in {0,16,32,48}: stays in-row
    unsigned ldsa = (unsigned)(uintptr_t)&Xs[r * XROW + c];
    const void* g = (const void*)(src + h0);
    asm volatile(
        "global_load_async_to_lds_b128 %0, %1, off\n\t"
        "global_load_async_to_lds_b128 %0, %1, off offset:16"
        :: "v"(ldsa), "v"(g) : "memory");
  }
  // ---- stage W block: 8192 contiguous floats, convert f32 -> f16 in VGPRs
  // (overlaps the async X transfer above)
  {
    const float* src = W + (size_t)p * (1024 * 64) + (size_t)nblk * (128 * 64);
    const int f0 = tid * 32;                 // 32 floats / thread
    const int r = f0 >> 6, c = f0 & 63;
    const float4* g = (const float4*)(src + f0);
#pragma unroll
    for (int q = 0; q < 4; ++q) {
      float4 a = g[2 * q], b = g[2 * q + 1];
      h8 h;
      h[0] = (_Float16)a.x; h[1] = (_Float16)a.y;
      h[2] = (_Float16)a.z; h[3] = (_Float16)a.w;
      h[4] = (_Float16)b.x; h[5] = (_Float16)b.y;
      h[6] = (_Float16)b.z; h[7] = (_Float16)b.w;
      *(h8*)&Ws[r * WROW + c + q * 8] = h;
    }
  }
  if (tid < 128) Bs[tid] = bvec[(size_t)p * 1024 + nblk * 128 + tid];

  // Each wave drains its own async transfers, then the barrier makes all
  // waves' LDS writes (async + ds_store) visible block-wide.
  asm volatile("s_wait_asynccnt 0x0" ::: "memory");
  __syncthreads();

  const int mt  = wave >> 1;          // 0..3
  const int ntb = (wave & 1) * 4;     // 0 or 4

  // C/D layout: VGPR r holds D[m = r + 8*half][n = l16]; bias depends on n only.
  v8f acc[4];
#pragma unroll
  for (int j = 0; j < 4; ++j) {
    float bv = Bs[(ntb + j) * 16 + l16];
#pragma unroll
    for (int r = 0; r < 8; ++r) acc[j][r] = bv;
  }

#pragma unroll
  for (int kc = 0; kc < 2; ++kc) {
    // A frag (16-bit 16x32 layout): lane row m=l16 of tile mt,
    // halves cover K = kc*32 + {half*8..half*8+7, 16+half*8..16+half*8+7}
    union { v16h v; uint4 q[2]; } A;
    const int arow  = mt * 16 + l16;
    const int akoff = kc * 32 + half * 8;
    A.q[0] = *(const uint4*)&Xs[arow * XROW + akoff];
    A.q[1] = *(const uint4*)&Xs[arow * XROW + akoff + 16];
#pragma unroll
    for (int j = 0; j < 4; ++j) {
      // B frag: B = W^T (col-major) => per-lane 16 halves (two 8-half runs) of W row
      // n=l16, K = kc*32 + {half*8..half*8+7, 16+half*8..16+half*8+7}
      union { v16h v; uint4 q[2]; } Bf;
      const int brow  = (ntb + j) * 16 + l16;
      const int bkoff = kc * 32 + half * 8;
      Bf.q[0] = *(const uint4*)&Ws[brow * WROW + bkoff];
      Bf.q[1] = *(const uint4*)&Ws[brow * WROW + bkoff + 16];
      acc[j] = __builtin_amdgcn_wmma_f32_16x16x32_f16(
          false, A.v, false, Bf.v, (short)0, acc[j], false, false);
    }
  }

  // ---- pixel-shuffle scatter: o = (t, ui, uj); addr = b*64^3 + t*4096 +
  // (Si*4+ui)*64 + Sj*4+uj. (t,ui,uj) fixed per (j,lane); only b varies with r.
#pragma unroll
  for (int j = 0; j < 4; ++j) {
    const int o  = nblk * 128 + (ntb + j) * 16 + l16;
    const int t  = o >> 4;
    const int ui = (o >> 2) & 3;
    const int uj = o & 3;
    const size_t K = (size_t)t * 4096 + (size_t)(Si * 4 + ui) * 64 + (Sj * 4 + uj);
    const int mbase = mblk * 64 + mt * 16 + half * 8;
#pragma unroll
    for (int r = 0; r < 8; ++r) {
      out[(size_t)(mbase + r) * 262144 + K] = acc[j][r];
    }
  }
}

extern "C" void kernel_launch(void* const* d_in, const int* in_sizes, int n_in,
                              void* d_out, int out_size, void* d_ws, size_t ws_size,
                              hipStream_t stream) {
  const float* x = (const float*)d_in[0];  // (256, 64, 16, 16)
  const float* W = (const float*)d_in[1];  // (256, 1024, 64)
  const float* b = (const float*)d_in[2];  // (256, 1024)
  float* out = (float*)d_out;              // (256, 64, 64, 64)
  _Float16* xh = (_Float16*)d_ws;          // needs 256*16384*2 = 8 MiB scratch

  dim3 tgrid(16384 / 32, 256 / 32);        // (512, 8)
  dim3 tblk(32, 8);
  ups_transpose_cvt<<<tgrid, tblk, 0, stream>>>(x, xh);

  dim3 ggrid(8, 4, 256);                   // (nblk, mblk, pixel)
  ups_wmma_gemm<<<ggrid, 256, 0, stream>>>(xh, W, b, out);
}
